// VolumeRenderer_1133871366455
// MI455X (gfx1250) — matrix operand order repaired and checked by reading, not compiled, not run on hardware
//
#include <hip/hip_runtime.h>
#include <hip/hip_bf16.h>

typedef __attribute__((ext_vector_type(16))) _Float16 v16h;
typedef __attribute__((ext_vector_type(8)))  _Float16 v8h;
typedef __attribute__((ext_vector_type(8)))  float    v8f;

#define HDIM 256
#define NS 64
#define LSTRIDE 264          // padded LDS row stride in halves (528B = 4-bank skew)
#define NEAR_T 1.0f
#define FAR_T 6.0f
#define FAR_APPEND_T 1000.0f

// ---------------------------------------------------------------------------
// Pre-pass: convert W2 (f32, row-major (k,n)) -> W2h (f16, transposed (n,k)).
// 128 KB result is reused by all workgroups; it lives in L2 (192 MB on MI455X).
// ---------------------------------------------------------------------------
__global__ void w2_convert_kernel(const float* __restrict__ W2,
                                  _Float16* __restrict__ W2h) {
    int k = blockIdx.x;      // 0..255
    int n = threadIdx.x;     // 0..255
    W2h[n * HDIM + k] = (_Float16)W2[k * HDIM + n];
}

// ---------------------------------------------------------------------------
// Fused NeRF forward: one workgroup (256 threads = 8 wave32) per ray.
// M = 64 GEMM rows = the ray's 64 samples. Layer2 (256x256) via WMMA f16.
// ---------------------------------------------------------------------------
__global__ __launch_bounds__(256) void nerf_fused_kernel(
    const float* __restrict__ x_pix, const float* __restrict__ Kmat,
    const float* __restrict__ c2w,
    const float* __restrict__ W1, const float* __restrict__ b1,
    const _Float16* __restrict__ W2h, const float* __restrict__ b2,
    const float* __restrict__ W3, const float* __restrict__ b3,
    float* __restrict__ out_rgb, float* __restrict__ out_depth)
{
    extern __shared__ __align__(16) unsigned char smem_raw[];
    _Float16* hA     = (_Float16*)smem_raw;          // 64 x 264 halves (33792 B), reused as hB
    float*    s_sig  = (float*)(hA + NS * LSTRIDE);  // 64 floats
    float*    s_rgb  = s_sig + NS;                   // 64*3 floats

    const int tid   = threadIdx.x;
    const int lane  = tid & 31;
    const int wave  = tid >> 5;
    const int g     = blockIdx.x;        // global ray index over B*R
    const int batch = g >> 12;           // R = 4096 rays per batch

    // ---- ray setup (uniform per block; scalar loads) ----
    const float px = x_pix[g * 2 + 0], py = x_pix[g * 2 + 1];
    const float* Kb = Kmat + batch * 9;
    float a00 = Kb[0], a01 = Kb[1], a02 = Kb[2];
    float a10 = Kb[3], a11 = Kb[4], a12 = Kb[5];
    float a20 = Kb[6], a21 = Kb[7], a22 = Kb[8];
    float det = a00 * (a11 * a22 - a12 * a21)
              - a01 * (a10 * a22 - a12 * a20)
              + a02 * (a10 * a21 - a11 * a20);
    float id = 1.0f / det;
    float i00 =  (a11 * a22 - a12 * a21) * id;
    float i01 = -(a01 * a22 - a02 * a21) * id;
    float i02 =  (a01 * a12 - a02 * a11) * id;
    float i10 = -(a10 * a22 - a12 * a20) * id;
    float i11 =  (a00 * a22 - a02 * a20) * id;
    float i12 = -(a00 * a12 - a02 * a10) * id;
    float i20 =  (a10 * a21 - a11 * a20) * id;
    float i21 = -(a00 * a21 - a01 * a20) * id;
    float i22 =  (a00 * a11 - a01 * a10) * id;
    float dc0 = i00 * px + i01 * py + i02;
    float dc1 = i10 * px + i11 * py + i12;
    float dc2 = i20 * px + i21 * py + i22;
    const float* M = c2w + batch * 16;
    float rdx = M[0] * dc0 + M[1] * dc1 + M[2]  * dc2;
    float rdy = M[4] * dc0 + M[5] * dc1 + M[6]  * dc2;
    float rdz = M[8] * dc0 + M[9] * dc1 + M[10] * dc2;
    float rox = M[3], roy = M[7], roz = M[11];
    const float dzval = (FAR_T - NEAR_T) / (float)(NS - 1);

    // ---- layer 1: h1[s][n] = relu((ro + z_s*rd) . W1[:,n] + b1[n]) ----
    // Factored: c0 + z_s * c1  -> one FMA per value. Thread t owns column n=t.
    {
        const int n = tid;
        float w0 = W1[n], w1 = W1[HDIM + n], w2 = W1[2 * HDIM + n];
        float c0 = rox * w0 + roy * w1 + roz * w2 + b1[n];
        float c1 = rdx * w0 + rdy * w1 + rdz * w2;
        #pragma unroll 8
        for (int s = 0; s < NS; ++s) {
            float z = NEAR_T + dzval * (float)s;
            float h = fmaxf(c0 + z * c1, 0.0f);
            hA[s * LSTRIDE + n] = (_Float16)h;
        }
    }
    __syncthreads();

    // ---- layer 2: 64x256 = hA(64x256) @ W2(256x256), WMMA f16 -> f32 ----
    // Wave w: row half a = w&1 (M-tiles 2a,2a+1), col quarter b = w>>1 (4 N-tiles)
    const int aidx  = wave & 1;
    const int bidx  = wave >> 1;     // 0..3
    const int halfl = lane >> 4;     // 0/1 (half-wave)
    const int l15   = lane & 15;

    float b2c[4];
    #pragma unroll
    for (int j = 0; j < 4; ++j) b2c[j] = b2[(4 * bidx + j) * 16 + l15];

    v8f acc[2][4];
    #pragma unroll
    for (int i = 0; i < 2; ++i)
        #pragma unroll
        for (int j = 0; j < 4; ++j)
            #pragma unroll
            for (int v = 0; v < 8; ++v) acc[i][j][v] = 0.0f;

    for (int k = 0; k < 8; ++k) {          // K loop: 8 x 32
        // A fragments (ISA 16-bit A 16x32 layout): two ds_load_b128 per frag
        v16h af[2];
        #pragma unroll
        for (int i = 0; i < 2; ++i) {
            int row = (2 * aidx + i) * 16 + l15;
            int kb  = 32 * k + (halfl ? 8 : 0);
            v8h lo = *(const v8h*)(hA + row * LSTRIDE + kb);
            v8h hi = *(const v8h*)(hA + row * LSTRIDE + kb + 16);
            af[i] = __builtin_shufflevector(lo, hi,
                     0,1,2,3,4,5,6,7,8,9,10,11,12,13,14,15);
        }
        // B fragments (lane = column, K in ISA table order per lane) from cached global
        #pragma unroll
        for (int j = 0; j < 4; ++j) {
            int n  = (4 * bidx + j) * 16 + l15;
            int kb = 32 * k + (halfl ? 8 : 0);
            v8h lo = *(const v8h*)(W2h + n * HDIM + kb);
            v8h hi = *(const v8h*)(W2h + n * HDIM + kb + 16);
            v16h bf = __builtin_shufflevector(lo, hi,
                      0,1,2,3,4,5,6,7,8,9,10,11,12,13,14,15);
            acc[0][j] = __builtin_amdgcn_wmma_f32_16x16x32_f16(
                false, af[0], false, bf, (short)0, acc[0][j], false, false);
            acc[1][j] = __builtin_amdgcn_wmma_f32_16x16x32_f16(
                false, af[1], false, bf, (short)0, acc[1][j], false, false);
        }
    }
    __syncthreads();   // all waves done reading hA; safe to overwrite with h2

    // ---- epilogue: bias + relu, store h2 into LDS (reuse hA) ----
    // C/D layout: VGPR v, lane l -> M = v + 8*(l>>4), N = l&15
    _Float16* hB = hA;
    #pragma unroll
    for (int i = 0; i < 2; ++i)
        #pragma unroll
        for (int j = 0; j < 4; ++j) {
            int col = (4 * bidx + j) * 16 + l15;
            #pragma unroll
            for (int v = 0; v < 8; ++v) {
                int row = (2 * aidx + i) * 16 + v + 8 * halfl;
                float xv = acc[i][j][v] + b2c[j];
                hB[row * LSTRIDE + col] = (_Float16)fmaxf(xv, 0.0f);
            }
        }
    __syncthreads();

    // ---- layer 3: 256 -> 4 (VALU dot products; 1 (row, out) pair per thread)
    {
        int row = tid >> 2;
        int c   = tid & 3;
        float sum = b3[c];
        const _Float16* hp = hB + row * LSTRIDE;
        #pragma unroll 8
        for (int k2 = 0; k2 < HDIM; ++k2)
            sum += (float)hp[k2] * W3[k2 * 4 + c];
        if (c == 0) s_sig[row] = fmaxf(sum, 0.0f);
        else        s_rgb[row * 3 + (c - 1)] = 1.0f / (1.0f + __expf(-sum));
    }
    __syncthreads();

    // ---- volume integral: inclusive cumprod scan over 64 samples ----
    if (tid == 0) {
        float T = 1.0f, r0 = 0.f, r1 = 0.f, r2 = 0.f, dep = 0.f, aw = 0.f;
        for (int s = 0; s < NS; ++s) {
            float z     = NEAR_T + dzval * (float)s;
            float dist  = (s < NS - 1) ? dzval : (FAR_APPEND_T - FAR_T);
            float alpha = 1.0f - __expf(-s_sig[s] * dist);
            T *= (1.0f - alpha);          // inclusive cumprod (torch-matching)
            float w = alpha * T;
            r0 += w * s_rgb[s * 3 + 0];
            r1 += w * s_rgb[s * 3 + 1];
            r2 += w * s_rgb[s * 3 + 2];
            dep += w * z;
            aw  += w;
        }
        float bgc = 1.0f - aw;            // white background
        out_rgb[g * 3 + 0] = r0 + bgc;
        out_rgb[g * 3 + 1] = r1 + bgc;
        out_rgb[g * 3 + 2] = r2 + bgc;
        out_depth[g] = dep;
    }
}

extern "C" void kernel_launch(void* const* d_in, const int* in_sizes, int n_in,
                              void* d_out, int out_size, void* d_ws, size_t ws_size,
                              hipStream_t stream) {
    const float* x_pix = (const float*)d_in[0];
    const float* Kmat  = (const float*)d_in[1];
    const float* c2w   = (const float*)d_in[2];
    const float* W1    = (const float*)d_in[3];
    const float* b1    = (const float*)d_in[4];
    const float* W2    = (const float*)d_in[5];
    const float* b2    = (const float*)d_in[6];
    const float* W3    = (const float*)d_in[7];
    const float* b3    = (const float*)d_in[8];

    const int BR = in_sizes[0] / 2;      // B*R total rays (8192)

    // scratch: W2 transposed + converted to f16 (128 KB)
    _Float16* W2h = (_Float16*)d_ws;
    w2_convert_kernel<<<HDIM, HDIM, 0, stream>>>(W2, W2h);

    float* out       = (float*)d_out;
    float* out_rgb   = out;              // BR*3 floats
    float* out_depth = out + (size_t)BR * 3;

    size_t smem = (size_t)NS * LSTRIDE * sizeof(_Float16)   // hA / hB (33792 B)
                + (size_t)NS * 4 * sizeof(float);           // sigma + rgb (1024 B)

    nerf_fused_kernel<<<BR, 256, smem, stream>>>(
        x_pix, Kmat, c2w, W1, b1, W2h, b2, W3, b3, out_rgb, out_depth);
}
